// SegSelfAtt_62577673503756
// MI455X (gfx1250) — matrix operand order repaired and checked by reading, not compiled, not run on hardware
//
#include <hip/hip_runtime.h>
#include <hip/hip_bf16.h>

typedef _Float16 half_t;
typedef half_t v16h __attribute__((ext_vector_type(16)));
typedef half_t v8h  __attribute__((ext_vector_type(8)));
typedef float  v8f  __attribute__((ext_vector_type(8)));
typedef unsigned int u32x4 __attribute__((ext_vector_type(4)));
typedef int i32x8 __attribute__((ext_vector_type(8)));
typedef int i32x4 __attribute__((ext_vector_type(4)));

#define NSEQ 128
#define NDIM 256
#define NEGV (-1e10f)

// ---------------- LDS plan (one 128-token segment per workgroup) ----------
// Phase 0: TDM stages raw f32 H into the (dead) K/Vt region [131072,262144).
// Phase 1: H(f16) | Q(f16) | K(f16) | V^T(f16) | P(f16)      = 288 KB
// Phase 2 (reuses dead space): G_Att(f32) | L_Att(f32)        = 264 KB
struct S1 {
    half_t Hh[NSEQ][NDIM];   // 64 KB   [0,      65536)
    half_t Qh[NSEQ][NDIM];   // 64 KB   [65536, 131072)
    half_t Kh[NSEQ][NDIM];   // 64 KB   [131072,196608)
    half_t Vt[NDIM][NSEQ];   // 64 KB   [196608,262144)  V transposed (d-major)
    half_t P [NSEQ][NSEQ];   // 32 KB   [262144,294912)  per-wave prob rows
};
struct S2 {
    float G[NSEQ][NDIM + 8]; // 132 KB  [0,     135168)  overlaps H,Q,K (dead after scores)
    float L[NSEQ][NDIM + 8]; // 132 KB  [135168,270336)  overlaps K tail,Vt,P head (dead after barrier)
};
union __align__(128) SMem { S1 s1; S2 s2; };

// A-fragment (16x32 f16): lane m%16 = row; lane-half holds 16 contiguous K
__device__ __forceinline__ v16h load_a(const half_t* p) {
    v8h lo = *(const v8h*)(p);
    v8h hi = *(const v8h*)(p + 8);
    return __builtin_shufflevector(lo, hi, 0,1,2,3,4,5,6,7,8,9,10,11,12,13,14,15);
}
// B-fragment (32x16 f16): lane n%16 = column; lane-half holds 16 contiguous K
__device__ __forceinline__ v16h load_b(const half_t* p) {
    return *(const v16h*)(p);
}
__device__ __forceinline__ v8f wmma16(v16h a, v16h b, v8f c) {
    return __builtin_amdgcn_wmma_f32_16x16x32_f16(false, a, false, b, (short)0, c, false, false);
}

// Prep: Wq/Wk/Wv (f32, [d_in][d_out]) -> transposed f16 [mat][d_out][d_in] in d_ws.
__global__ void prep_weights_kernel(const float* __restrict__ Wq,
                                    const float* __restrict__ Wk,
                                    const float* __restrict__ Wv,
                                    half_t* __restrict__ Wt) {
    int i = blockIdx.x * 256 + threadIdx.x;     // 0 .. 3*65536-1
    int m = i >> 16;
    int r = i & 65535;
    int o = r >> 8;                              // output dim
    int k = r & 255;                             // input dim
    const float* W = (m == 0) ? Wq : (m == 1) ? Wk : Wv;
    Wt[i] = (half_t)W[k * NDIM + o];
}

__global__ void __launch_bounds__(256)
seg_attn_kernel(const float* __restrict__ Hs,
                const unsigned char* __restrict__ segm,   // (B,L,1,S) bool
                const half_t* __restrict__ Wt,            // [3][256][256] f16, transposed
                const float* __restrict__ Whw, const float* __restrict__ Whb,
                const float* __restrict__ Wgw, const float* __restrict__ Wlw,
                const float* __restrict__ lng, const float* __restrict__ lnb,
                const unsigned char* __restrict__ band,   // (1,1,S,S) bool
                float* __restrict__ out) {
    __shared__ SMem sm;
    const int seg  = blockIdx.x;
    const int tid  = threadIdx.x;
    const int lane = tid & 31;
    const int wave = tid >> 5;
    const int ln16 = lane & 15;
    const int hi   = lane >> 4;
    const int m0   = wave * 16;          // this wave's 16 query rows (attention stages)
    const int arow = m0 + ln16;          // A-fragment row for this lane
    const int rb   = m0 + hi * 8;        // C-fragment row base (elems r -> rb+r)

    const float* Hseg = Hs + (size_t)seg * NSEQ * NDIM;

    // ---- Stage 1: H (f32 global) -> Hh (f16 LDS) -------------------------
#if __has_builtin(__builtin_amdgcn_tensor_load_to_lds) && __has_builtin(__builtin_amdgcn_s_wait_tensorcnt)
    {
        float* Hstg = (float*)&sm.s1.Kh[0][0];   // 128 KB staging in dead K/Vt space
        if (wave == 0) {
            // Tensor DMA: one 1-D tile of 32768 f32 elements, global -> LDS.
            unsigned long long ga = (unsigned long long)(uintptr_t)Hseg;
            unsigned int ldsa = (unsigned int)(uintptr_t)Hstg;   // flat low 32 = LDS byte addr
            u32x4 g0;
            g0[0] = 1u;                                           // count=1, user load
            g0[1] = ldsa;                                         // lds_addr
            g0[2] = (unsigned int)ga;                             // global_addr[31:0]
            g0[3] = (unsigned int)((ga >> 32) & 0x1FFFFFFull)     // global_addr[56:32]
                    | (2u << 30);                                 // type=2 (image/tensor)
            i32x8 g1;
            g1[0] = (int)(2u << 16);          // data_size=4B
            g1[1] = (int)(32768u << 16);      // tensor_dim0[15:0] = 32768
            g1[2] = (int)(1u << 16);          // tensor_dim0[31:16]=0 | tensor_dim1=1
            g1[3] = (int)(32768u << 16);      // tile_dim0 = 32768
            g1[4] = 0;                        // tile_dim1=0 (unused), tile_dim2=0
            g1[5] = 32768;                    // tensor_dim0_stride[31:0]
            g1[6] = 0;
            g1[7] = 0;
            i32x4 g2 = {0,0,0,0};
            i32x4 g3 = {0,0,0,0};
            i32x8 g4 = {0,0,0,0,0,0,0,0};     // trailing group (clang-23 6-arg form)
            __builtin_amdgcn_tensor_load_to_lds(g0, g1, g2, g3, g4, 0);
            __builtin_amdgcn_s_wait_tensorcnt(0);
        }
        __syncthreads();
        // all waves: convert staged f32 -> f16 Hh
        for (int it = 0; it < 32; ++it) {
            int i4 = tid + it * 256;                     // 8192 float4 total
            float4 h = ((const float4*)Hstg)[i4];
            half_t* d = &sm.s1.Hh[0][0] + i4 * 4;
            d[0] = (half_t)h.x; d[1] = (half_t)h.y; d[2] = (half_t)h.z; d[3] = (half_t)h.w;
        }
    }
#else
    for (int it = 0; it < 32; ++it) {
        int i4 = tid + it * 256;
        float4 h = ((const float4*)Hseg)[i4];
        half_t* d = &sm.s1.Hh[0][0] + i4 * 4;
        d[0] = (half_t)h.x; d[1] = (half_t)h.y; d[2] = (half_t)h.z; d[3] = (half_t)h.w;
    }
#endif
    __syncthreads();

    // ---- Stage 2: Q = H Wq / 16, K = H Wk, V^T ---------------------------
    // Each wave owns 2 output N-tiles (32 dims) for ALL 128 rows -> the block
    // reads each weight matrix from L2 exactly once (B-fragments stay in VGPRs).
    {
        const int nbase = wave * 32;
        #pragma unroll 1
        for (int mat = 0; mat < 3; ++mat) {
            const half_t* Wm = Wt + mat * (NDIM * NDIM);
            v16h bf[2][8];
            #pragma unroll
            for (int nt = 0; nt < 2; ++nt) {
                const half_t* wrow = Wm + (nbase + nt * 16 + ln16) * NDIM + hi * 16;
                #pragma unroll
                for (int kb = 0; kb < 8; ++kb) bf[nt][kb] = load_b(wrow + kb * 32);
            }
            #pragma unroll 1
            for (int rt = 0; rt < 8; ++rt) {
                v16h a[8];
                #pragma unroll
                for (int kb = 0; kb < 8; ++kb)
                    a[kb] = load_a(&sm.s1.Hh[rt * 16 + ln16][kb * 32 + hi * 16]);
                const int rowb = rt * 16 + hi * 8;
                #pragma unroll
                for (int nt = 0; nt < 2; ++nt) {
                    const int n = nbase + nt * 16 + ln16;
                    v8f acc = {0,0,0,0,0,0,0,0};
                    #pragma unroll
                    for (int kb = 0; kb < 8; ++kb) acc = wmma16(a[kb], bf[nt][kb], acc);
                    if (mat == 0) {
                        #pragma unroll
                        for (int r = 0; r < 8; ++r)
                            sm.s1.Qh[rowb + r][n] = (half_t)(acc[r] * 0.0625f);
                    } else if (mat == 1) {
                        #pragma unroll
                        for (int r = 0; r < 8; ++r)
                            sm.s1.Kh[rowb + r][n] = (half_t)acc[r];
                    } else {
                        v8h pk;
                        #pragma unroll
                        for (int r = 0; r < 8; ++r) pk[r] = (half_t)acc[r];
                        *(v8h*)&sm.s1.Vt[n][rowb] = pk;       // transposed, 16B store
                    }
                }
            }
        }
    }
    __syncthreads();

    // ---- Stage 3: scores S = (Q/16) K^T  (kept in registers) -------------
    v16h af[8];
    #pragma unroll
    for (int kb = 0; kb < 8; ++kb) af[kb] = load_a(&sm.s1.Qh[arow][kb * 32 + hi * 16]);
    v8f sc[8];
    #pragma unroll
    for (int t = 0; t < 8; ++t) {
        const half_t* krow = &sm.s1.Kh[t * 16 + ln16][hi * 16];
        v8f acc = {0,0,0,0,0,0,0,0};
        #pragma unroll
        for (int kb = 0; kb < 8; ++kb) acc = wmma16(af[kb], load_b(krow + kb * 32), acc);
        sc[t] = acc;
    }
    const unsigned char* smk = segm + seg * NSEQ;
    #pragma unroll
    for (int t = 0; t < 8; ++t) {
        if (!smk[t * 16 + ln16]) {
            #pragma unroll
            for (int r = 0; r < 8; ++r) sc[t][r] = NEGV;
        }
    }
    __syncthreads();   // Hh/Qh/Kh dead everywhere; safe to overlay s2.G

    // ---- softmax over register scores, write f16 probs to own P rows -----
    auto softmax_and_store = [&](v8f (&s)[8]) {
        float vmax[8], vsum[8];
        #pragma unroll
        for (int r = 0; r < 8; ++r) {
            float m = s[0][r];
            #pragma unroll
            for (int t = 1; t < 8; ++t) m = fmaxf(m, s[t][r]);
            vmax[r] = m;
        }
        #pragma unroll
        for (int d = 1; d < 16; d <<= 1)
            #pragma unroll
            for (int r = 0; r < 8; ++r) vmax[r] = fmaxf(vmax[r], __shfl_xor(vmax[r], d, 32));
        #pragma unroll
        for (int r = 0; r < 8; ++r) vsum[r] = 0.f;
        #pragma unroll
        for (int t = 0; t < 8; ++t)
            #pragma unroll
            for (int r = 0; r < 8; ++r) {
                float e = __expf(s[t][r] - vmax[r]);
                s[t][r] = e; vsum[r] += e;
            }
        #pragma unroll
        for (int d = 1; d < 16; d <<= 1)
            #pragma unroll
            for (int r = 0; r < 8; ++r) vsum[r] += __shfl_xor(vsum[r], d, 32);
        #pragma unroll
        for (int r = 0; r < 8; ++r) vsum[r] = 1.f / vsum[r];
        #pragma unroll
        for (int t = 0; t < 8; ++t) {
            int n = t * 16 + ln16;
            #pragma unroll
            for (int r = 0; r < 8; ++r) sm.s1.P[rb + r][n] = (half_t)(s[t][r] * vsum[r]);
        }
    };

    // ---- Stage 4: GLOBAL path  G_Att = softmax(S) V  -> s2.G -------------
    {
        v8f g[8];
        #pragma unroll
        for (int t = 0; t < 8; ++t) g[t] = sc[t];
        softmax_and_store(g);
        v16h pf[4];
        #pragma unroll
        for (int kb = 0; kb < 4; ++kb) pf[kb] = load_a(&sm.s1.P[arow][kb * 32 + hi * 16]);
        #pragma unroll 1
        for (int t = 0; t < 16; ++t) {
            const half_t* vrow = &sm.s1.Vt[t * 16 + ln16][hi * 16];
            v8f acc = {0,0,0,0,0,0,0,0};
            #pragma unroll
            for (int kb = 0; kb < 4; ++kb) acc = wmma16(pf[kb], load_b(vrow + kb * 32), acc);
            int n = t * 16 + ln16;
            #pragma unroll
            for (int r = 0; r < 8; ++r) sm.s2.G[rb + r][n] = acc[r];
        }
    }

    // ---- Stage 5: LOCAL path  L_Att = softmax(band ? S : NEG) V ----------
    {
        #pragma unroll
        for (int t = 0; t < 8; ++t) {
            int n = t * 16 + ln16;
            #pragma unroll
            for (int r = 0; r < 8; ++r)
                if (!band[(rb + r) * NSEQ + n]) sc[t][r] = NEGV;
        }
        softmax_and_store(sc);
        v16h pf[4];
        #pragma unroll
        for (int kb = 0; kb < 4; ++kb) pf[kb] = load_a(&sm.s1.P[arow][kb * 32 + hi * 16]);
        v8f lac[16];
        #pragma unroll
        for (int t = 0; t < 16; ++t) {
            const half_t* vrow = &sm.s1.Vt[t * 16 + ln16][hi * 16];
            v8f acc = {0,0,0,0,0,0,0,0};
            #pragma unroll
            for (int kb = 0; kb < 4; ++kb) acc = wmma16(pf[kb], load_b(vrow + kb * 32), acc);
            lac[t] = acc;
        }
        __syncthreads();            // all waves done reading Vt/P -> overlay s2.L
        #pragma unroll
        for (int t = 0; t < 16; ++t) {
            int n = t * 16 + ln16;
            #pragma unroll
            for (int r = 0; r < 8; ++r) sm.s2.L[rb + r][n] = lac[t][r];
        }
    }
    __syncthreads();

    // ---- Stage 6: gate + fusion + residual + LayerNorm (float4) ----------
    {
        const int row = tid >> 1;
        const int c0  = (tid & 1) * 128;              // lane^1 is the partner
        const float4* H4  = (const float4*)(Hseg + row * NDIM + c0);
        const float4* G4  = (const float4*)&sm.s2.G[row][c0];
        float4*       L4  = (float4*)&sm.s2.L[row][c0];
        const float4* Wh4 = (const float4*)(Whw + c0);
        const float4* Wg4 = (const float4*)(Wgw + c0);
        const float4* Wl4 = (const float4*)(Wlw + c0);

        float hd = 0.f, gd = 0.f, ld = 0.f;
        for (int c = 0; c < 32; ++c) {
            float4 h = H4[c], g = G4[c], l = L4[c];
            float4 wh = Wh4[c], wg = Wg4[c], wl = Wl4[c];
            hd += h.x*wh.x + h.y*wh.y + h.z*wh.z + h.w*wh.w;
            gd += g.x*wg.x + g.y*wg.y + g.z*wg.z + g.w*wg.w;
            ld += l.x*wl.x + l.y*wl.y + l.z*wl.z + l.w*wl.w;
        }
        hd += __shfl_xor(hd, 1, 32);
        gd += __shfl_xor(gd, 1, 32);
        ld += __shfl_xor(ld, 1, 32);
        float z = hd + Whb[0] + gd + ld;
        float gate = 1.f / (1.f + __expf(-z));

        float sum = 0.f, ss = 0.f;
        for (int c = 0; c < 32; ++c) {
            float4 h = H4[c], g = G4[c], l = L4[c], x;
            x.x = gate*l.x + (1.f-gate)*g.x + h.x;
            x.y = gate*l.y + (1.f-gate)*g.y + h.y;
            x.z = gate*l.z + (1.f-gate)*g.z + h.z;
            x.w = gate*l.w + (1.f-gate)*g.w + h.w;
            L4[c] = x;                                 // stash fused value
            sum += x.x + x.y + x.z + x.w;
            ss  += x.x*x.x + x.y*x.y + x.z*x.z + x.w*x.w;
        }
        sum += __shfl_xor(sum, 1, 32);
        ss  += __shfl_xor(ss, 1, 32);
        float mu  = sum * (1.f / 256.f);
        float var = ss * (1.f / 256.f) - mu * mu;
        float inv = rsqrtf(var + 1e-6f);

        float4* O4 = (float4*)(out + ((size_t)seg * NSEQ + row) * NDIM + c0);
        const float4* Lg4 = (const float4*)(lng + c0);
        const float4* Lb4 = (const float4*)(lnb + c0);
        for (int c = 0; c < 32; ++c) {
            float4 x = L4[c], g = Lg4[c], b = Lb4[c], o;
            o.x = g.x * (x.x - mu) * inv + b.x;
            o.y = g.y * (x.y - mu) * inv + b.y;
            o.z = g.z * (x.z - mu) * inv + b.z;
            o.w = g.w * (x.w - mu) * inv + b.w;
            O4[c] = o;
        }
    }
}

extern "C" void kernel_launch(void* const* d_in, const int* in_sizes, int n_in,
                              void* d_out, int out_size, void* d_ws, size_t ws_size,
                              hipStream_t stream) {
    (void)in_sizes; (void)n_in; (void)out_size; (void)ws_size;
    const float*         Hs   = (const float*)d_in[0];
    const unsigned char* segm = (const unsigned char*)d_in[1];   // bool array
    const float*         Wq   = (const float*)d_in[2];
    const float*         Wk   = (const float*)d_in[3];
    const float*         Wv   = (const float*)d_in[4];
    const float*         Whw  = (const float*)d_in[5];
    const float*         Whb  = (const float*)d_in[6];
    const float*         Wgw  = (const float*)d_in[7];
    const float*         Wlw  = (const float*)d_in[8];
    const float*         lng  = (const float*)d_in[9];
    const float*         lnb  = (const float*)d_in[10];
    const unsigned char* band = (const unsigned char*)d_in[11];  // bool array

    half_t* Wt = (half_t*)d_ws;   // 3 * 256 * 256 f16 = 384 KB

    prep_weights_kernel<<<768, 256, 0, stream>>>(Wq, Wk, Wv, Wt);
    seg_attn_kernel<<<512, 256, 0, stream>>>(Hs, segm, Wt, Whw, Whb, Wgw, Wlw,
                                             lng, lnb, band, (float*)d_out);
}
